// OrientationAligner_33492154974287
// MI455X (gfx1250) — matrix operand order repaired and checked by reading, not compiled, not run on hardware
//
#include <hip/hip_runtime.h>
#include <math.h>

// ---------------- types for CDNA5 WMMA ----------------
typedef _Float16 half_t;
typedef __attribute__((ext_vector_type(16))) _Float16 v16h;
typedef __attribute__((ext_vector_type(8)))  _Float16 v8h;
typedef __attribute__((ext_vector_type(8)))  float    v8f;

union U16H { v16h v; v8h h8[2]; _Float16 e[16]; };

#define BATCH 16
#define NPTS  4096
#define BN3   (BATCH*NPTS*3)      // 196608
#define PI_4  0.7853981633974483f // 2*pi/8 == pi/4; lookup[k]=k*PI_4 - PI_4

// workspace layout (float offsets)
#define OFF_CENS   0
#define OFF_CENT   48
#define OFF_ANG    96
#define OFF_SRCAUG 128
#define OFF_TGTAUG (OFF_SRCAUG + BN3)            // 196736
#define OFF_PART   (OFF_TGTAUG + BN3)            // 393344  [4][16][8][256]
#define OFF_G      (OFF_PART + 4*16*8*256)       // 524416  [4][16][256]

#define ENC_SMEM  158208
#define HEAD_SMEM 66816

__device__ __forceinline__ v8f wmma32(v16h a, v16h b, v8f c) {
  // D = A(16x32 f16) * B(32x16 f16) + C(16x16 f32)
  return __builtin_amdgcn_wmma_f32_16x16x32_f16(false, a, false, b, (short)0, c,
                                                false, false);
}

// ---------------- 1) per-batch centroids ----------------
__global__ void oa_centroid_kernel(const float* __restrict__ src,
                                   const float* __restrict__ tgt,
                                   float* __restrict__ ws) {
  __shared__ float red[256];
  int bb = blockIdx.x;            // 0..31: 0-15 source, 16-31 target
  int b = bb & 15;
  const float* p = (bb < 16) ? src : tgt;
  float* out = ws + ((bb < 16) ? OFF_CENS : OFF_CENT) + b * 3;
  float sums[3] = {0.f, 0.f, 0.f};
  for (int i = threadIdx.x; i < NPTS; i += 256) {
    const float* q = p + ((size_t)b * NPTS + i) * 3;
    sums[0] += q[0]; sums[1] += q[1]; sums[2] += q[2];
  }
  for (int d = 0; d < 3; ++d) {
    red[threadIdx.x] = sums[d];
    __syncthreads();
    for (int off = 128; off > 0; off >>= 1) {
      if (threadIdx.x < off) red[threadIdx.x] += red[threadIdx.x + off];
      __syncthreads();
    }
    if (threadIdx.x == 0) out[d] = red[0] * (1.0f / NPTS);
    __syncthreads();
  }
}

// ---------------- 2) augmentation ----------------
__global__ void oa_augment_kernel(const float* __restrict__ src,
                                  const float* __restrict__ tgt,
                                  const float* __restrict__ ns,
                                  const float* __restrict__ nt,
                                  const int* __restrict__ rgt,
                                  float* __restrict__ ws) {
  int i = blockIdx.x * 256 + threadIdx.x;
  if (i >= BATCH * NPTS) return;
  int b = i >> 12;
  size_t o = (size_t)i * 3;
  float* srcAug = ws + OFF_SRCAUG;
  float* tgtAug = ws + OFF_TGTAUG;
  srcAug[o + 0] = src[o + 0] + ns[o + 0] * 0.01f;
  srcAug[o + 1] = src[o + 1] + ns[o + 1] * 0.01f;
  srcAug[o + 2] = src[o + 2] + ns[o + 2] * 0.01f;

  const float* cen = ws + OFF_CENT + b * 3;
  float ang = (float)rgt[b] * PI_4 - PI_4;
  float c = cosf(ang), s = sinf(ang);
  float px = tgt[o + 0] - cen[0];
  float py = tgt[o + 1] - cen[1];
  float pz = tgt[o + 2] - cen[2];
  tgtAug[o + 0] = px * c - pz * s + cen[0] + nt[o + 0] * 0.01f;
  tgtAug[o + 1] = py            + cen[1] + nt[o + 1] * 0.01f;
  tgtAug[o + 2] = px * s + pz * c + cen[2] + nt[o + 2] * 0.01f;
}

// ---------------- 3) WMMA encoder: shared MLP + maxpool ----------------
// grid.x = 4 clouds * 16 batches * 8 chunks = 512; 256 threads = 8 waves.
// Each wave processes 2x16-point tiles per iteration so every weight
// B-fragment ds_load feeds two v_wmma ops.
// dynamic LDS layout (bytes):
//   sW2T  half[128][64]     @ 0        (W2 transposed, n-major)
//   sW3T  half[256][128]    @ 16384
//   sW1   float[3][64]      @ 81920
//   sb1   float[64]         @ 82688
//   sb2   float[128]        @ 82944
//   sb3   float[256]        @ 83456
//   sh2   half[16][16][128] @ 84480    (per (wave,subtile) h2 staging)
//   sred  float[8][256]     @ 150016
__global__ void __launch_bounds__(256)
oa_encode_kernel(const float* __restrict__ c0, const float* __restrict__ c1,
                 const float* __restrict__ c2, const float* __restrict__ c3,
                 const float* __restrict__ W1, const float* __restrict__ b1,
                 const float* __restrict__ W2, const float* __restrict__ b2,
                 const float* __restrict__ W3, const float* __restrict__ b3,
                 float* __restrict__ partial) {
  extern __shared__ char smem[];
  half_t* sW2T = (half_t*)(smem);
  half_t* sW3T = (half_t*)(smem + 16384);
  float*  sW1  = (float*)(smem + 81920);
  float*  sb1  = (float*)(smem + 82688);
  float*  sb2  = (float*)(smem + 82944);
  float*  sb3  = (float*)(smem + 83456);
  half_t* sh2  = (half_t*)(smem + 84480);
  float*  sred = (float*)(smem + 150016);

  int tid = threadIdx.x;
  int blk = blockIdx.x;
  int cloud_id = blk >> 7;
  int rem = blk & 127;
  int b = rem >> 3;
  int chunk = rem & 7;
  const float* cloud = (cloud_id == 0) ? c0 : (cloud_id == 1) ? c1
                      : (cloud_id == 2) ? c2 : c3;

  // stage weights: f16 convert + transpose (n-major), float4 global loads
  {
    const float4* W2v = (const float4*)W2;
    for (int i = tid; i < (64 * 128) / 4; i += 256) {
      float4 w = W2v[i];
      int k = i >> 5, n = (i & 31) * 4;
      sW2T[(n + 0) * 64 + k] = (half_t)w.x;
      sW2T[(n + 1) * 64 + k] = (half_t)w.y;
      sW2T[(n + 2) * 64 + k] = (half_t)w.z;
      sW2T[(n + 3) * 64 + k] = (half_t)w.w;
    }
    const float4* W3v = (const float4*)W3;
    for (int i = tid; i < (128 * 256) / 4; i += 256) {
      float4 w = W3v[i];
      int k = i >> 6, n = (i & 63) * 4;
      sW3T[(n + 0) * 128 + k] = (half_t)w.x;
      sW3T[(n + 1) * 128 + k] = (half_t)w.y;
      sW3T[(n + 2) * 128 + k] = (half_t)w.z;
      sW3T[(n + 3) * 128 + k] = (half_t)w.w;
    }
  }
  if (tid < 192) sW1[tid] = W1[tid];
  if (tid < 64)  sb1[tid] = b1[tid];
  if (tid < 128) sb2[tid] = b2[tid];
  sb3[tid] = b3[tid];
  __syncthreads();

  int lane = tid & 31, wave = tid >> 5;
  int lo = lane & 15, hi = lane >> 4;

  float rmax[16];
#pragma unroll
  for (int t = 0; t < 16; ++t) rmax[t] = 0.f;

  for (int it = 0; it < 2; ++it) {
    // ---- layer 1 (K=3) into A-fragment layout for both subtiles ----
    U16H a1[2][2];
#pragma unroll
    for (int u = 0; u < 2; ++u) {
      int tile = wave + 8 * u + 16 * it;
      int pt = chunk * 512 + tile * 16 + lo;
      const float* xp = cloud + ((size_t)b * NPTS + pt) * 3;
      float x0 = xp[0], x1 = xp[1], x2 = xp[2];
#pragma unroll
      for (int s = 0; s < 2; ++s)
#pragma unroll
        for (int h = 0; h < 16; ++h) {
          int k = 32 * s + ((h >> 3) << 4) + (hi << 3) + (h & 7);
          float v = sb1[k] + x0 * sW1[k] + x1 * sW1[64 + k] + x2 * sW1[128 + k];
          a1[u][s].e[h] = (half_t)fmaxf(v, 0.f);
        }
    }

    // ---- layer 2: each B fragment feeds both subtiles ----
#pragma unroll
    for (int t = 0; t < 8; ++t) {
      float bv = sb2[t * 16 + lo];
      v8f acc[2];
#pragma unroll
      for (int r = 0; r < 8; ++r) { acc[0][r] = bv; acc[1][r] = bv; }
#pragma unroll
      for (int s = 0; s < 2; ++s) {
        U16H bf;
        const v8h* bp = (const v8h*)(sW2T + (t * 16 + lo) * 64 + 32 * s + hi * 8);
        bf.h8[0] = bp[0];
        bf.h8[1] = bp[2];
        acc[0] = wmma32(a1[0][s].v, bf.v, acc[0]);
        acc[1] = wmma32(a1[1][s].v, bf.v, acc[1]);
      }
#pragma unroll
      for (int u = 0; u < 2; ++u)
#pragma unroll
        for (int r = 0; r < 8; ++r) {
          int m = r + hi * 8;
          sh2[((wave * 2 + u) * 16 + m) * 128 + t * 16 + lo] =
              (half_t)fmaxf(acc[u][r], 0.f);
        }
    }
    __syncthreads();

    // ---- hoist layer-3 A fragments (k-step invariant across N-tiles) ----
    U16H a3[2][4];
#pragma unroll
    for (int u = 0; u < 2; ++u)
#pragma unroll
      for (int s = 0; s < 4; ++s) {
        int base = ((wave * 2 + u) * 16 + lo) * 128 + 32 * s + hi * 8;
        a3[u][s].h8[0] = *(const v8h*)(sh2 + base);
        a3[u][s].h8[1] = *(const v8h*)(sh2 + base + 16);
      }

    // ---- layer 3: 16 N-tiles x 4 K-steps, fused relu + maxpool ----
#pragma unroll
    for (int t3 = 0; t3 < 16; ++t3) {
      float bv = sb3[t3 * 16 + lo];
      v8f acc[2];
#pragma unroll
      for (int r = 0; r < 8; ++r) { acc[0][r] = bv; acc[1][r] = bv; }
#pragma unroll
      for (int s = 0; s < 4; ++s) {
        U16H bf;
        const v8h* bp = (const v8h*)(sW3T + (t3 * 16 + lo) * 128 + 32 * s + hi * 8);
        bf.h8[0] = bp[0];
        bf.h8[1] = bp[2];
        acc[0] = wmma32(a3[0][s].v, bf.v, acc[0]);
        acc[1] = wmma32(a3[1][s].v, bf.v, acc[1]);
      }
      float mx = rmax[t3];
#pragma unroll
      for (int r = 0; r < 8; ++r) {
        mx = fmaxf(mx, acc[0][r]);   // relu folded: rmax starts at 0
        mx = fmaxf(mx, acc[1][r]);
      }
      rmax[t3] = mx;
    }
    __syncthreads();
  }

  // wave-level reduce: lanes [0,16) and [16,32) hold same channel
  for (int t3 = 0; t3 < 16; ++t3)
    if (hi == 0) sred[wave * 256 + t3 * 16 + lo] = rmax[t3];
  __syncthreads();
  for (int t3 = 0; t3 < 16; ++t3)
    if (hi == 1) {
      float* p = sred + wave * 256 + t3 * 16 + lo;
      *p = fmaxf(*p, rmax[t3]);
    }
  __syncthreads();

  float m = sred[tid];
#pragma unroll
  for (int w = 1; w < 8; ++w) m = fmaxf(m, sred[w * 256 + tid]);
  partial[(((size_t)cloud_id * 16 + b) * 8 + chunk) * 256 + tid] = m;
}

// ---------------- 4) reduce chunk partials ----------------
__global__ void oa_reducemax_kernel(const float* __restrict__ partial,
                                    float* __restrict__ g) {
  int cb = blockIdx.x;  // (cloud*16+b), 0..63
  int j = threadIdx.x;
  const float* p = partial + (size_t)cb * 8 * 256 + j;
  float m = p[0];
  for (int c = 1; c < 8; ++c) m = fmaxf(m, p[c * 256]);
  g[(size_t)cb * 256 + j] = m;
}

// ---------------- 5) head MLP + losses ----------------
__global__ void __launch_bounds__(256)
oa_head_kernel(const float* __restrict__ g,
               const float* __restrict__ Wm1, const float* __restrict__ bm1,
               const float* __restrict__ Wm2, const float* __restrict__ bm2,
               const float* __restrict__ Wm3, const float* __restrict__ bm3,
               const float* __restrict__ Wa, const float* __restrict__ ba,
               const float* __restrict__ Wd, const float* __restrict__ bd,
               const int* __restrict__ rgt, float* __restrict__ angOut,
               float* __restrict__ out) {
  extern __shared__ char smem[];
  float* h  = (float*)smem;            // 16*512
  float* m1 = (float*)(smem + 32768);  // 16*256
  float* m2 = (float*)(smem + 49152);  // 16*128
  float* m3 = (float*)(smem + 57344);  // 16*128
  float* al = (float*)(smem + 65536);  // [2][16][8]
  float* dl = (float*)(smem + 66560);  // [2][16][2]
  int t = threadIdx.x;

  for (int pair = 0; pair < 2; ++pair) {
    const float* gS = g + (size_t)(pair * 2) * 16 * 256;
    const float* gT = g + (size_t)(pair * 2 + 1) * 16 * 256;
    for (int i = t; i < 16 * 512; i += 256) {
      int b = i >> 9, j = i & 511;
      h[i] = (j < 256) ? gS[b * 256 + j] : gT[b * 256 + (j - 256)];
    }
    __syncthreads();
    {
      int o = t;
      for (int b = 0; b < 16; ++b) {
        float acc = bm1[o];
        for (int k = 0; k < 512; ++k) acc += h[b * 512 + k] * Wm1[k * 256 + o];
        m1[b * 256 + o] = fmaxf(acc, 0.f);
      }
    }
    __syncthreads();
    if (t < 128) {
      int o = t;
      for (int b = 0; b < 16; ++b) {
        float acc = bm2[o];
        for (int k = 0; k < 256; ++k) acc += m1[b * 256 + k] * Wm2[k * 128 + o];
        m2[b * 128 + o] = fmaxf(acc, 0.f);
      }
    }
    __syncthreads();
    if (t < 128) {
      int o = t;
      for (int b = 0; b < 16; ++b) {
        float acc = bm3[o];
        for (int k = 0; k < 128; ++k) acc += m2[b * 128 + k] * Wm3[k * 128 + o];
        m3[b * 128 + o] = fmaxf(acc, 0.f);
      }
    }
    __syncthreads();
    if (t < 128) {
      int b = t >> 3, j = t & 7;
      float acc = ba[j];
      for (int k = 0; k < 128; ++k) acc += m3[b * 128 + k] * Wa[k * 8 + j];
      al[pair * 128 + b * 8 + j] = acc;
    }
    if (t < 32) {
      int b = t >> 1, j = t & 1;
      float acc = bd[j];
      for (int k = 0; k < 128; ++k) acc += m3[b * 128 + k] * Wd[k * 2 + j];
      dl[pair * 32 + b * 2 + j] = acc;
    }
    __syncthreads();
  }

  if (t == 0) {
    // align angles from clean logits (first max wins, matches argmax)
    for (int b = 0; b < 16; ++b) {
      int best = 0;
      float bv = al[b * 8];
      for (int j = 1; j < 8; ++j) {
        float v = al[b * 8 + j];
        if (v > bv) { bv = v; best = j; }
      }
      angOut[b] = -((float)best * PI_4 - PI_4);
    }
    // cross-entropy on augmented angle logits
    float aloss = 0.f;
    for (int b = 0; b < 16; ++b) {
      const float* row = al + 128 + b * 8;
      float mx = row[0];
      for (int j = 1; j < 8; ++j) mx = fmaxf(mx, row[j]);
      float se = 0.f;
      for (int j = 0; j < 8; ++j) se += expf(row[j] - mx);
      aloss += (mx + logf(se)) - row[rgt[b]];
    }
    aloss = (aloss / 16.f) * 0.4f;
    // focal losses: clean->label 0, aug->label 1
    float dloss = 0.f;
    for (int pass = 0; pass < 2; ++pass) {
      float s = 0.f;
      for (int b = 0; b < 16; ++b) {
        const float* row = dl + pass * 32 + b * 2;
        float mx = fmaxf(row[0], row[1]);
        float lse = mx + logf(expf(row[0] - mx) + expf(row[1] - mx));
        float pt = expf(row[pass] - lse);
        float om = 1.f - pt;
        s += -(om * om * om) * logf(pt + 1e-8f);
      }
      dloss += s / 16.f;
    }
    out[2 * BN3 + 0] = aloss;
    out[2 * BN3 + 1] = dloss;
  }
}

// ---------------- 6) align source + copy target ----------------
__global__ void oa_align_kernel(const float* __restrict__ src,
                                const float* __restrict__ tgt,
                                const float* __restrict__ cenS,
                                const float* __restrict__ angA,
                                float* __restrict__ out) {
  int i = blockIdx.x * 256 + threadIdx.x;
  if (i >= BATCH * NPTS) return;
  int b = i >> 12;
  size_t o = (size_t)i * 3;
  float ang = angA[b];
  float c = cosf(ang), s = sinf(ang);
  const float* cen = cenS + b * 3;
  float px = src[o + 0] - cen[0];
  float py = src[o + 1] - cen[1];
  float pz = src[o + 2] - cen[2];
  out[o + 0] = px * c - pz * s + cen[0];
  out[o + 1] = py + cen[1];
  out[o + 2] = px * s + pz * c + cen[2];
  out[BN3 + o + 0] = tgt[o + 0];
  out[BN3 + o + 1] = tgt[o + 1];
  out[BN3 + o + 2] = tgt[o + 2];
}

extern "C" void kernel_launch(void* const* d_in, const int* in_sizes, int n_in,
                              void* d_out, int out_size, void* d_ws, size_t ws_size,
                              hipStream_t stream) {
  (void)in_sizes; (void)n_in; (void)out_size; (void)ws_size;
  const float* source    = (const float*)d_in[0];
  const float* target    = (const float*)d_in[1];
  const float* noise_src = (const float*)d_in[2];
  const float* noise_tgt = (const float*)d_in[3];
  const int*   rgt       = (const int*)d_in[4];
  const float* W1 = (const float*)d_in[5];  const float* b1 = (const float*)d_in[6];
  const float* W2 = (const float*)d_in[7];  const float* b2 = (const float*)d_in[8];
  const float* W3 = (const float*)d_in[9];  const float* b3 = (const float*)d_in[10];
  const float* Wm1 = (const float*)d_in[11]; const float* bm1 = (const float*)d_in[12];
  const float* Wm2 = (const float*)d_in[13]; const float* bm2 = (const float*)d_in[14];
  const float* Wm3 = (const float*)d_in[15]; const float* bm3 = (const float*)d_in[16];
  const float* Wa = (const float*)d_in[17]; const float* ba = (const float*)d_in[18];
  const float* Wd = (const float*)d_in[19]; const float* bd = (const float*)d_in[20];
  float* ws  = (float*)d_ws;
  float* out = (float*)d_out;

  oa_centroid_kernel<<<32, 256, 0, stream>>>(source, target, ws);
  oa_augment_kernel<<<256, 256, 0, stream>>>(source, target, noise_src, noise_tgt,
                                             rgt, ws);
  oa_encode_kernel<<<512, 256, ENC_SMEM, stream>>>(
      source, target, ws + OFF_SRCAUG, ws + OFF_TGTAUG,
      W1, b1, W2, b2, W3, b3, ws + OFF_PART);
  oa_reducemax_kernel<<<64, 256, 0, stream>>>(ws + OFF_PART, ws + OFF_G);
  oa_head_kernel<<<1, 256, HEAD_SMEM, stream>>>(
      ws + OFF_G, Wm1, bm1, Wm2, bm2, Wm3, bm3, Wa, ba, Wd, bd,
      rgt, ws + OFF_ANG, out);
  oa_align_kernel<<<256, 256, 0, stream>>>(source, target, ws + OFF_CENS,
                                           ws + OFF_ANG, out);
}
